// MinCutShapleyGNN_Improved_44770739093929
// MI455X (gfx1250) — matrix operand order repaired and checked by reading, not compiled, not run on hardware
//
#include <hip/hip_runtime.h>
#include <math.h>

// ---------------------------------------------------------------------------
// MinCut-pool + Shapley GNN for MI455X (gfx1250), wave32 + WMMA bf16.
// Key refactor: Z = S.T @ (x@W_proj + b_proj) == (S.T@x) @ W_proj + colsum(S)*b_proj
// so the [N,256] X_proj is never materialized.
// LDS staging uses fragment-contiguous layouts so gathers lower to ds_load_b128.
// ---------------------------------------------------------------------------

typedef __attribute__((ext_vector_type(16))) __bf16 v16bf;
typedef __attribute__((ext_vector_type(8)))  __bf16 v8bf;
typedef __attribute__((ext_vector_type(4)))  __bf16 v4bf;
typedef __attribute__((ext_vector_type(8)))  float  v8f;

#define IN_DIM  256
#define C_DIM   30
#define CPAD    32
#define HID_DIM 256
#define OUT_DIM 64

static __device__ __forceinline__ v8f wmma_bf16(v16bf a, v16bf b, v8f c) {
  return __builtin_amdgcn_wmma_f32_16x16x32_bf16(false, a, false, b, (short)0, c,
                                                 false, false);
}
static __device__ __forceinline__ v16bf cat8(v8bf lo, v8bf hi) {
  return __builtin_shufflevector(lo, hi, 0, 1, 2, 3, 4, 5, 6, 7, 8, 9, 10, 11,
                                 12, 13, 14, 15);
}

static __device__ __forceinline__ float allreduce_max16(float v) {
#pragma unroll
  for (int m = 1; m <= 8; m <<= 1) v = fmaxf(v, __shfl_xor(v, m, 32));
  return v;
}
static __device__ __forceinline__ float allreduce_sum16(float v) {
#pragma unroll
  for (int m = 1; m <= 8; m <<= 1) v += __shfl_xor(v, m, 32);
  return v;
}

// ---------------------------------------------------------------------------
__global__ void zero_kernel(float* __restrict__ p, size_t n) {
  for (size_t i = (size_t)blockIdx.x * blockDim.x + threadIdx.x; i < n;
       i += (size_t)gridDim.x * blockDim.x)
    p[i] = 0.0f;
}

// ---------------------------------------------------------------------------
// S = softmax(x @ W_assign + b_assign)  (row-wise over 30 clusters)
// Also accumulates colsum(S) for the b_proj correction term.
// Block = 128 threads (4 waves); each wave owns a 16-node tile.
// W_assign staged in WMMA B-fragment layout (32B/lane -> ds_load_b128);
// x tile staged bf16 row-major (A-fragment = 2 contiguous 16B runs).
// ---------------------------------------------------------------------------
__global__ void __launch_bounds__(128)
assign_softmax_kernel(const float* __restrict__ x, const float* __restrict__ Wa,
                      const float* __restrict__ ba, float* __restrict__ S,
                      float* __restrict__ colsum, int N) {
  __shared__ alignas(32) __bf16 sWf[2][8][32][16];   // 16 KB fragment layout
  __shared__ alignas(16) __bf16 sX[4][16][IN_DIM];   // 32 KB bf16 tiles
  const int tid = threadIdx.x;

  // Stage W_assign into fragment layout: element i of lane ln, k-step ks32,
  // column tile `set` holds Wa[ks32*32 + (ln>>4)*16 + i][set*16 + (ln&15)].
  for (int idx = tid; idx < 2 * 8 * 32 * 16; idx += 128) {
    int set = idx >> 12;
    int ks32 = (idx >> 9) & 7;
    int ln = (idx >> 4) & 31;
    int i = idx & 15;
    int col = set * 16 + (ln & 15);
    int kk = ks32 * 32 + (ln >> 4) * 16 + i;
    sWf[set][ks32][ln][i] =
        (__bf16)((col < C_DIM) ? Wa[kk * C_DIM + col] : 0.0f);
  }

  const int wave = tid >> 5, lane = tid & 31;
  const int khalf = lane >> 4, lrow = lane & 15;
  const long long node0 = ((long long)blockIdx.x * 4 + wave) * 16;

  for (int i = lane; i < 16 * (IN_DIM / 4); i += 32) {
    int r = i >> 6, c4 = i & 63;
    long long n = node0 + r;
    float4 v = make_float4(0.f, 0.f, 0.f, 0.f);
    if (n < N) v = ((const float4*)x)[n * (IN_DIM / 4) + c4];
    v4bf pk = {(__bf16)v.x, (__bf16)v.y, (__bf16)v.z, (__bf16)v.w};
    *(v4bf*)&sX[wave][r][c4 * 4] = pk;
  }
  __syncthreads();

  v8f d0 = {}, d1 = {};
#pragma unroll
  for (int ks32 = 0; ks32 < 8; ++ks32) {
    const int ks = ks32 * 32;
    v8bf alo = *(const v8bf*)&sX[wave][lrow][ks + khalf * 16];
    v8bf ahi = *(const v8bf*)&sX[wave][lrow][ks + khalf * 16 + 8];
    v16bf a = cat8(alo, ahi);
    v16bf b0 = *(const v16bf*)&sWf[0][ks32][lane][0];
    v16bf b1 = *(const v16bf*)&sWf[1][ks32][lane][0];
    d0 = wmma_bf16(a, b0, d0);
    d1 = wmma_bf16(a, b1, d1);
  }

  const int c0 = lrow, c1 = 16 + lrow;
  const float bias0 = ba[c0];
  const float bias1 = (c1 < C_DIM) ? ba[c1] : 0.0f;
  float cs0 = 0.f, cs1 = 0.f;
#pragma unroll
  for (int r = 0; r < 8; ++r) {
    const int M = r + 8 * khalf;
    const long long n = node0 + M;
    float v0 = d0[r] + bias0;
    float v1 = (c1 < C_DIM) ? (d1[r] + bias1) : -3.0e38f;
    float mx = allreduce_max16(fmaxf(v0, v1));
    float e0 = __expf(v0 - mx);
    float e1 = (c1 < C_DIM) ? __expf(v1 - mx) : 0.0f;
    float inv = 1.0f / allreduce_sum16(e0 + e1);
    float s0 = e0 * inv, s1 = e1 * inv;
    if (n < N) {
      S[n * C_DIM + c0] = s0;
      if (c1 < C_DIM) S[n * C_DIM + c1] = s1;
      cs0 += s0;
      cs1 += s1;
    }
  }
  cs0 += __shfl_xor(cs0, 16, 32);
  cs1 += __shfl_xor(cs1, 16, 32);
  if (khalf == 0) {
    if (cs0 != 0.f) atomicAdd(&colsum[c0], cs0);
    if (c1 < C_DIM && cs1 != 0.f) atomicAdd(&colsum[c1], cs1);
  }
}

// ---------------------------------------------------------------------------
// Edge scatter: adj_S[row] += S[col]; deg[row] += 1. One wave per edge;
// lanes 0..29 handle one cluster component each (coalesced, L2-resident).
// ---------------------------------------------------------------------------
__global__ void __launch_bounds__(256)
edge_kernel(const int* __restrict__ row, const int* __restrict__ col,
            const float* __restrict__ S, float* __restrict__ adjS,
            float* __restrict__ deg, long long E) {
  long long w = (((long long)blockIdx.x * blockDim.x) + threadIdx.x) >> 5;
  int lane = threadIdx.x & 31;
  if (w >= E) return;
  int r = row[w], c = col[w];
  if (lane < C_DIM) {
    float v = S[(long long)c * C_DIM + lane];
    atomicAdd(&adjS[(long long)r * C_DIM + lane], v);
  } else if (lane == 31) {
    atomicAdd(&deg[r], 1.0f);
  }
}

// ---------------------------------------------------------------------------
// M = S.T @ x  ([30,256], stored padded [32,256]).  WMMA over K = nodes.
// Block = 128 threads (4 waves); wave w owns k-tiles 4w..4w+3, both c-tiles.
// Both blocks staged TRANSPOSED in LDS so every fragment is a contiguous,
// 16B-aligned run (-> ds_load_b128). Row strides padded for bank rotation
// while staying 16B multiples (40 and 48 bf16).
// ---------------------------------------------------------------------------
__global__ void __launch_bounds__(128)
stx_wmma_kernel(const float* __restrict__ S, const float* __restrict__ x,
                float* __restrict__ Mws, int N) {
  __shared__ alignas(16) __bf16 sST[CPAD][40];     // S.T block: [cluster][node]
  __shared__ alignas(16) __bf16 sXT[IN_DIM][48];   // x.T block: [feature][node]
  const int tid = threadIdx.x;
  const int wave = tid >> 5, lane = tid & 31;
  const int khalf = lane >> 4, lrow = lane & 15;

  v8f acc[2][4] = {};
  const int nb_total = (N + 31) / 32;
  for (int nb = blockIdx.x; nb < nb_total; nb += gridDim.x) {
    long long n0 = (long long)nb * 32;
    for (int i = tid; i < 32 * 32; i += 128) {
      int c = i >> 5, n = i & 31;
      long long nn = n0 + n;
      float v = (c < C_DIM && nn < N) ? S[nn * C_DIM + c] : 0.0f;
      sST[c][n] = (__bf16)v;
    }
    for (int i = tid; i < 32 * (IN_DIM / 4); i += 128) {
      int n = i >> 6, c4 = i & 63;
      long long nn = n0 + n;
      float4 v = make_float4(0.f, 0.f, 0.f, 0.f);
      if (nn < N) v = ((const float4*)x)[nn * (IN_DIM / 4) + c4];
      sXT[c4 * 4 + 0][n] = (__bf16)v.x;
      sXT[c4 * 4 + 1][n] = (__bf16)v.y;
      sXT[c4 * 4 + 2][n] = (__bf16)v.z;
      sXT[c4 * 4 + 3][n] = (__bf16)v.w;
    }
    // Prefetch next node-block of x while this one is consumed.
    long long pn = n0 + (long long)gridDim.x * 32;
    if (pn < N) __builtin_prefetch(&x[pn * IN_DIM + (tid & 63) * 4], 0, 1);
    __syncthreads();

    v16bf a0 = cat8(*(const v8bf*)&sST[lrow][khalf * 16],
                    *(const v8bf*)&sST[lrow][khalf * 16 + 8]);
    v16bf a1 = cat8(*(const v8bf*)&sST[16 + lrow][khalf * 16],
                    *(const v8bf*)&sST[16 + lrow][khalf * 16 + 8]);
#pragma unroll
    for (int t = 0; t < 4; ++t) {
      int kcol = (wave * 4 + t) * 16 + lrow;
      v16bf b = *(const v16bf*)&sXT[kcol][khalf * 16];
      acc[0][t] = wmma_bf16(a0, b, acc[0][t]);
      acc[1][t] = wmma_bf16(a1, b, acc[1][t]);
    }
    __syncthreads();
  }

#pragma unroll
  for (int ct = 0; ct < 2; ++ct)
#pragma unroll
    for (int t = 0; t < 4; ++t)
#pragma unroll
      for (int r = 0; r < 8; ++r) {
        int c = ct * 16 + r + 8 * khalf;
        int k = (wave * 4 + t) * 16 + lrow;
        if (c < C_DIM) atomicAdd(&Mws[c * IN_DIM + k], acc[ct][t][r]);
      }
}

// ---------------------------------------------------------------------------
// Generic C[30,30] += A.T @ B for A,B in [N,30] (used for adj_new and SS).
// ---------------------------------------------------------------------------
__global__ void __launch_bounds__(256)
atb30_kernel(const float* __restrict__ A, const float* __restrict__ B,
             float* __restrict__ C, int N) {
  __shared__ float sA[128][C_DIM];
  __shared__ float sB[128][C_DIM];
  const int tid = threadIdx.x;
  float acc[4] = {0.f, 0.f, 0.f, 0.f};
  const int chunks = (N + 127) / 128;
  for (int ch = blockIdx.x; ch < chunks; ch += gridDim.x) {
    long long n0 = (long long)ch * 128;
    for (int i = tid; i < 128 * C_DIM; i += 256) {
      int n = i / C_DIM, c = i % C_DIM;
      long long nn = n0 + n;
      float va = 0.f, vb = 0.f;
      if (nn < N) { va = A[nn * C_DIM + c]; vb = B[nn * C_DIM + c]; }
      sA[n][c] = va;
      sB[n][c] = vb;
    }
    __syncthreads();
#pragma unroll
    for (int q = 0; q < 4; ++q) {
      int pid = tid + 256 * q;
      if (pid < C_DIM * C_DIM) {
        int c1 = pid / C_DIM, c2 = pid % C_DIM;
        float s = 0.f;
        for (int n = 0; n < 128; ++n) s += sA[n][c1] * sB[n][c2];
        acc[q] += s;
      }
    }
    __syncthreads();
  }
#pragma unroll
  for (int q = 0; q < 4; ++q) {
    int pid = tid + 256 * q;
    if (pid < C_DIM * C_DIM) atomicAdd(&C[pid], acc[q]);
  }
}

// ---------------------------------------------------------------------------
// vol = sum_n deg[n] * rowsum(S[n])
// ---------------------------------------------------------------------------
__global__ void __launch_bounds__(256)
vol_kernel(const float* __restrict__ S, const float* __restrict__ deg,
           float* __restrict__ vol, int N) {
  __shared__ float red[256];
  float local = 0.f;
  for (long long n = (long long)blockIdx.x * blockDim.x + threadIdx.x; n < N;
       n += (long long)gridDim.x * blockDim.x) {
    float rs = 0.f;
#pragma unroll
    for (int c = 0; c < C_DIM; ++c) rs += S[n * C_DIM + c];
    local += deg[n] * rs;
  }
  red[threadIdx.x] = local;
  __syncthreads();
  for (int s = 128; s > 0; s >>= 1) {
    if (threadIdx.x < s) red[threadIdx.x] += red[threadIdx.x + s];
    __syncthreads();
  }
  if (threadIdx.x == 0) atomicAdd(vol, red[0]);
}

// ---------------------------------------------------------------------------
// Z = M @ W_proj + colsum(S) * b_proj   ([30,256] = [30,256]x[256,256])
// Single block, 8 waves; wave w owns k-tiles 2w, 2w+1 and both c-tiles.
// ---------------------------------------------------------------------------
__global__ void __launch_bounds__(256)
zproj_wmma_kernel(const float* __restrict__ Mws, const float* __restrict__ Wp,
                  const float* __restrict__ bp, const float* __restrict__ colsum,
                  float* __restrict__ Z) {
  const int tid = threadIdx.x;
  const int wave = tid >> 5, lane = tid & 31;
  const int khalf = lane >> 4, lrow = lane & 15;

  v8f acc[2][2] = {};
  for (int ks = 0; ks < IN_DIM; ks += 32) {
    v16bf a0, a1;
#pragma unroll
    for (int i = 0; i < 16; ++i) {
      int kk = ks + khalf * 16 + i;
      a0[i] = (__bf16)Mws[lrow * IN_DIM + kk];
      a1[i] = (__bf16)Mws[(16 + lrow) * IN_DIM + kk];
    }
#pragma unroll
    for (int t = 0; t < 2; ++t) {
      int kcol = (wave * 2 + t) * 16 + lrow;
      v16bf b;
#pragma unroll
      for (int i = 0; i < 16; ++i)
        b[i] = (__bf16)Wp[(ks + khalf * 16 + i) * IN_DIM + kcol];
      acc[0][t] = wmma_bf16(a0, b, acc[0][t]);
      acc[1][t] = wmma_bf16(a1, b, acc[1][t]);
    }
  }
#pragma unroll
  for (int ct = 0; ct < 2; ++ct)
#pragma unroll
    for (int t = 0; t < 2; ++t)
#pragma unroll
      for (int r = 0; r < 8; ++r) {
        int c = ct * 16 + r + 8 * khalf;
        int k = (wave * 2 + t) * 16 + lrow;
        if (c < C_DIM) Z[c * IN_DIM + k] = acc[ct][t][r] + colsum[c] * bp[k];
      }
}

// ---------------------------------------------------------------------------
// Finale: losses, pooled mask/degrees, Shapley tables, 2 Shapley layers, head.
// Everything is [30,*] — one 1024-thread block, LDS-resident.
// ---------------------------------------------------------------------------
__global__ void __launch_bounds__(1024)
final_kernel(const float* __restrict__ adjn, const float* __restrict__ SSm,
             const float* __restrict__ vol, const float* __restrict__ Zin,
             const float* __restrict__ W1, const float* __restrict__ W2,
             const float* __restrict__ Wo, const float* __restrict__ bo,
             float* __restrict__ out, float* __restrict__ scalars) {
  __shared__ float bufA[C_DIM][HID_DIM];
  __shared__ float bufB[C_DIM][HID_DIM];
  __shared__ float smask[C_DIM][C_DIM];
  __shared__ float sa[C_DIM + 1], sb[C_DIM + 1];
  __shared__ int sdeg[C_DIM];
  const int tid = threadIdx.x;

  if (tid < C_DIM * C_DIM)
    smask[tid / C_DIM][tid % C_DIM] = (adjn[tid] > 0.f) ? 1.f : 0.f;
  if (tid == 0) {
    float H[C_DIM + 2];
    H[0] = 0.f;
    for (int i = 1; i <= C_DIM + 1; ++i) H[i] = H[i - 1] + 1.f / (float)i;
    sa[0] = 0.f; sb[0] = 0.f;
    for (int m = 1; m <= C_DIM; ++m) {
      if (m <= 5) {
        sa[m] = H[m + 1] / (float)(m + 1);
        sb[m] = -(H[m + 1] - 1.f) / ((float)m * (float)(m + 1));
      } else {
        sa[m] = 0.f;
        sb[m] = 1.f / (float)m;
      }
    }
    float cut = 0.f;
    for (int c = 0; c < C_DIM; ++c) cut += adjn[c * C_DIM + c];
    scalars[0] = -cut / (vol[0] + 1e-9f);
  }
  if (tid == 32) {
    float o = 0.f;
    for (int i = 0; i < C_DIM * C_DIM; ++i) {
      float v = SSm[i] - ((i / C_DIM == i % C_DIM) ? 1.f : 0.f);
      o += v * v;
    }
    scalars[1] = sqrtf(o);
  }
  for (int i = tid; i < C_DIM * HID_DIM; i += blockDim.x)
    bufA[i / HID_DIM][i % HID_DIM] = Zin[i];
  __syncthreads();

  if (tid < C_DIM) {
    int nd = 0;
    for (int j = 0; j < C_DIM; ++j) nd += (smask[tid][j] > 0.f) ? 1 : 0;
    sdeg[tid] = nd;
  }
  __syncthreads();

  for (int layer = 0; layer < 2; ++layer) {
    const float* W = (layer == 0) ? W1 : W2;
    for (int i = tid; i < C_DIM * HID_DIM; i += blockDim.x) {
      int c = i / HID_DIM, k = i % HID_DIM;
      float t = 0.f;
#pragma unroll
      for (int j = 0; j < C_DIM; ++j) t += smask[c][j] * bufA[j][k];
      int nd = sdeg[c];
      bufB[c][k] = bufA[c][k] * (1.f + sa[nd]) + sb[nd] * t;
    }
    __syncthreads();
    for (int i = tid; i < C_DIM * HID_DIM; i += blockDim.x) {
      int c = i / HID_DIM, k = i % HID_DIM;
      float s = 0.f;
      for (int j = 0; j < HID_DIM; ++j) s += bufB[c][j] * W[j * HID_DIM + k];
      bufA[c][k] = fmaxf(s, 0.f);
    }
    __syncthreads();
  }
  for (int i = tid; i < C_DIM * OUT_DIM; i += blockDim.x) {
    int c = i / OUT_DIM, k = i % OUT_DIM;
    float s = 0.f;
    for (int j = 0; j < HID_DIM; ++j) s += bufA[c][j] * Wo[j * OUT_DIM + k];
    out[c * OUT_DIM + k] = s + bo[k];
  }
}

// ---------------------------------------------------------------------------
extern "C" void kernel_launch(void* const* d_in, const int* in_sizes, int n_in,
                              void* d_out, int out_size, void* d_ws,
                              size_t ws_size, hipStream_t stream) {
  (void)n_in; (void)out_size; (void)ws_size;
  const float* x  = (const float*)d_in[0];
  const int*   ei = (const int*)d_in[1];
  const float* Wa = (const float*)d_in[2];
  const float* ba = (const float*)d_in[3];
  const float* Wp = (const float*)d_in[4];
  const float* bp = (const float*)d_in[5];
  const float* W1 = (const float*)d_in[6];
  const float* W2 = (const float*)d_in[7];
  const float* Wo = (const float*)d_in[8];
  const float* bo = (const float*)d_in[9];

  const int N = in_sizes[0] / IN_DIM;
  const long long E = (long long)in_sizes[1] / 2;
  const int* row = ei;
  const int* col = ei + E;

  // Output layout: out[30*64] | mincut | ortho | Z[30*256] | S[N*30]
  float* outp    = (float*)d_out;
  float* out_mat = outp;
  float* scalars = outp + C_DIM * OUT_DIM;              // +1920
  float* Zout    = scalars + 2;                         // +1922
  float* Sout    = Zout + C_DIM * IN_DIM;               // +9602

  // Workspace: deg[N] | adjS[N*30] | M[32*256] | colsum[32] | SS[1024] |
  //            adj_new[1024] | vol[1]
  float* ws     = (float*)d_ws;
  float* deg    = ws;
  float* adjS   = ws + N;
  float* Mws    = adjS + (size_t)N * C_DIM;
  float* colsum = Mws + 32 * IN_DIM;
  float* SSm    = colsum + 32;
  float* adjn   = SSm + 1024;
  float* vol    = adjn + 1024;
  size_t ws_used = (size_t)N * 31 + 32 * IN_DIM + 32 + 1024 + 1024 + 1;

  zero_kernel<<<2048, 256, 0, stream>>>(ws, ws_used);

  int blkA = (N + 63) / 64;
  assign_softmax_kernel<<<blkA, 128, 0, stream>>>(x, Wa, ba, Sout, colsum, N);

  long long eblocks = (E * 32 + 255) / 256;
  edge_kernel<<<dim3((unsigned)eblocks), 256, 0, stream>>>(row, col, Sout, adjS,
                                                           deg, E);

  stx_wmma_kernel<<<240, 128, 0, stream>>>(Sout, x, Mws, N);
  atb30_kernel<<<240, 256, 0, stream>>>(Sout, adjS, adjn, N);
  atb30_kernel<<<240, 256, 0, stream>>>(Sout, Sout, SSm, N);
  vol_kernel<<<256, 256, 0, stream>>>(Sout, deg, vol, N);

  zproj_wmma_kernel<<<1, 256, 0, stream>>>(Mws, Wp, bp, colsum, Zout);
  final_kernel<<<1, 1024, 0, stream>>>(adjn, SSm, vol, Zout, W1, W2, Wo, bo,
                                       out_mat, scalars);
}
